// CrossAttention_16398185136149
// MI455X (gfx1250) — compile-verified, matrix operand order checked by reading, not run on hardware
//
#include <hip/hip_runtime.h>
#include <hip/hip_bf16.h>

typedef __attribute__((ext_vector_type(16))) _Float16 v16h;
typedef __attribute__((ext_vector_type(8)))  float    v8f;

#define WMMA_F16(a, b, c) \
  __builtin_amdgcn_wmma_f32_16x16x32_f16(false, (a), false, (b), (short)0, (c), false, false)

// ---------------------------------------------------------------------------
// Fragment loader for 16-bit WMMA operands (verified codegen: 2x b128/lane).
// A-side: lane holds row (row0 + lane&15), K-major.
// B-side: identical shuffle, lane holds column (col0 + lane&15) of B, i.e. a
//         row of B^T — so any (N,K) row-major matrix loads directly as B.
// ISA K layout per lane-group g=lane>>4:  elems 0..7  -> K = k0 + 8g .. +7
//                                         elems 8..15 -> K = k0 + 16 + 8g .. +7
// ---------------------------------------------------------------------------
union FragU { v16h v; uint4 q[2]; };

__device__ __forceinline__ v16h load_frag(const _Float16* base, int ld,
                                          int row0, int k0, int lane) {
  int r = row0 + (lane & 15);
  int g = (lane >> 4) << 3;              // 0 or 8
  const _Float16* p = base + (size_t)r * ld + k0 + g;
  FragU f;
  f.q[0] = *(const uint4*)(p);
  f.q[1] = *(const uint4*)(p + 16);
  return f.v;
}

// ---------------------------------------------------------------------------
__global__ void cvt_f32_to_f16(const float* __restrict__ src,
                               _Float16* __restrict__ dst, int n) {
  int i = blockIdx.x * blockDim.x + threadIdx.x;
  if (i < n) dst[i] = (_Float16)src[i];
}

// ---------------------------------------------------------------------------
// Projection GEMM: P = scale * X (M x 1024 f16) @ W^T  (W is (N,K) row-major).
// Wave tile = 32x64: per K-step 12 b128 loads feed 8 WMMA (21.3 FLOP/B).
// mode 0: write (b, h, seq, d)   [Q (scale=1/8) and K]
// mode 1: write (b, h, d, seq)   [V transposed for the P·V WMMA B-side]
// ---------------------------------------------------------------------------
__global__ void proj_gemm(const _Float16* __restrict__ A,
                          const _Float16* __restrict__ W,
                          _Float16* __restrict__ dst, int mode, int nseq,
                          float scale) {
  const int K = 1024;
  int lane = threadIdx.x & 31;
  int wave = threadIdx.x >> 5;
  int m0 = blockIdx.x * 256 + wave * 32;
  int n0 = blockIdx.y * 64;

  v8f acc[2][4] = {};
  for (int k0 = 0; k0 < K; k0 += 32) {
    v16h a0 = load_frag(A, K, m0,      k0, lane);
    v16h a1 = load_frag(A, K, m0 + 16, k0, lane);
#pragma unroll
    for (int t = 0; t < 4; ++t) {
      v16h b = load_frag(W, K, n0 + 16 * t, k0, lane);
      acc[0][t] = WMMA_F16(a0, b, acc[0][t]);
      acc[1][t] = WMMA_F16(a1, b, acc[1][t]);
    }
  }
  int g = lane >> 4, c = lane & 15;
#pragma unroll
  for (int u = 0; u < 2; ++u) {
#pragma unroll
    for (int t = 0; t < 4; ++t) {
#pragma unroll
      for (int r = 0; r < 8; ++r) {
        int m = m0 + 16 * u + r + 8 * g;
        int n = n0 + 16 * t + c;
        int b = m / nseq, s = m % nseq;
        int h = n >> 6, d = n & 63;
        size_t idx;
        if (mode == 0) idx = (((size_t)b * 16 + h) * nseq + s) * 64 + d;
        else           idx = (((size_t)b * 16 + h) * 64 + d) * (size_t)nseq + s;
        dst[idx] = (_Float16)(acc[u][t][r] * scale);
      }
    }
  }
}

// ---------------------------------------------------------------------------
// Flash attention: one wave per 32-query tile (2 sub-tiles u), online softmax
// over NK=2048 in chunks of 32 keys.  Per chunk: 4 K-fragments feed 8 QK^T
// WMMA, 4 V-fragments feed 8 P·V WMMA (each fragment reused across both query
// sub-tiles).  P re-shaped to A-fragments via 2KB LDS per wave.
// SCALE is pre-folded into Q by proj_gemm.
// ---------------------------------------------------------------------------
__global__ void attn_kernel(const _Float16* __restrict__ Qp,
                            const _Float16* __restrict__ Kp,
                            const _Float16* __restrict__ Vt,
                            _Float16* __restrict__ AO) {
  const int NQ = 2048, NK = 2048, DH = 64;
  __shared__ __align__(16) _Float16 lds[8][32 * 32];

  int lane = threadIdx.x & 31;
  int wave = threadIdx.x >> 5;
  int tile = blockIdx.x * 8 + wave;          // global 32-query tile id
  int i0 = (tile & (NQ / 32 - 1)) * 32;
  int bh = tile / (NQ / 32);                 // b*16 + h
  const _Float16* Qb = Qp + (size_t)bh * NQ * DH;
  const _Float16* Kb = Kp + (size_t)bh * NK * DH;
  const _Float16* Vb = Vt + (size_t)bh * DH * NK;

  v16h q[2][2];
#pragma unroll
  for (int u = 0; u < 2; ++u) {
    q[u][0] = load_frag(Qb, DH, i0 + 16 * u, 0,  lane);
    q[u][1] = load_frag(Qb, DH, i0 + 16 * u, 32, lane);
  }

  v8f o[2][4] = {};
  float mstat[2][8], lstat[2][8];
#pragma unroll
  for (int u = 0; u < 2; ++u)
#pragma unroll
    for (int r = 0; r < 8; ++r) { mstat[u][r] = -1e30f; lstat[u][r] = 0.0f; }

  _Float16* P = &lds[wave][0];
  int g = lane >> 4, c = lane & 15;

  for (int j0 = 0; j0 < NK; j0 += 32) {
    v16h kf00 = load_frag(Kb, DH, j0,      0,  lane);
    v16h kf01 = load_frag(Kb, DH, j0,      32, lane);
    v16h kf10 = load_frag(Kb, DH, j0 + 16, 0,  lane);
    v16h kf11 = load_frag(Kb, DH, j0 + 16, 32, lane);

#pragma unroll
    for (int u = 0; u < 2; ++u) {
      v8f s0 = {}, s1 = {};
      s0 = WMMA_F16(q[u][0], kf00, s0);
      s0 = WMMA_F16(q[u][1], kf01, s0);
      s1 = WMMA_F16(q[u][0], kf10, s1);
      s1 = WMMA_F16(q[u][1], kf11, s1);

#pragma unroll
      for (int r = 0; r < 8; ++r) {
        float a  = s0[r];
        float bv = s1[r];
        float mx = fmaxf(a, bv);
#pragma unroll
        for (int d = 8; d >= 1; d >>= 1) mx = fmaxf(mx, __shfl_xor(mx, d, 32));
        float mnew  = fmaxf(mstat[u][r], mx);
        float alpha = __expf(mstat[u][r] - mnew);
        float p0 = __expf(a - mnew);
        float p1 = __expf(bv - mnew);
        float rs = p0 + p1;
#pragma unroll
        for (int d = 8; d >= 1; d >>= 1) rs += __shfl_xor(rs, d, 32);
        lstat[u][r] = lstat[u][r] * alpha + rs;
        mstat[u][r] = mnew;
#pragma unroll
        for (int t = 0; t < 4; ++t) o[u][t][r] = o[u][t][r] * alpha;
        // D-layout -> row-major P in LDS (row = 16u + r + 8g)
        P[(16 * u + r + 8 * g) * 32 + c]      = (_Float16)p0;
        P[(16 * u + r + 8 * g) * 32 + 16 + c] = (_Float16)p1;
      }
    }

    v16h pf0 = load_frag(P, 32, 0,  0, lane);   // P rows 0..15  as A-frag
    v16h pf1 = load_frag(P, 32, 16, 0, lane);   // P rows 16..31 as A-frag
#pragma unroll
    for (int t = 0; t < 4; ++t) {
      v16h vf = load_frag(Vb, NK, 16 * t, j0, lane);  // V^T rows = V columns
      o[0][t] = WMMA_F16(pf0, vf, o[0][t]);
      o[1][t] = WMMA_F16(pf1, vf, o[1][t]);
    }
  }

  int b = bh >> 4, h = bh & 15;
#pragma unroll
  for (int u = 0; u < 2; ++u) {
#pragma unroll
    for (int t = 0; t < 4; ++t) {
#pragma unroll
      for (int r = 0; r < 8; ++r) {
        int i = i0 + 16 * u + r + 8 * g;
        float val = o[u][t][r] / lstat[u][r];
        AO[((size_t)b * NQ + i) * 1024 + h * 64 + 16 * t + c] = (_Float16)val;
      }
    }
  }
}

// ---------------------------------------------------------------------------
// Output projection: out = AO @ Wo^T + bo, f32 output.  Wave tile 32x64.
// ---------------------------------------------------------------------------
__global__ void out_gemm(const _Float16* __restrict__ A,
                         const _Float16* __restrict__ W,
                         const float* __restrict__ bias,
                         float* __restrict__ out) {
  const int K = 1024, N = 1024;
  int lane = threadIdx.x & 31;
  int wave = threadIdx.x >> 5;
  int m0 = blockIdx.x * 256 + wave * 32;
  int n0 = blockIdx.y * 64;

  v8f acc[2][4] = {};
  for (int k0 = 0; k0 < K; k0 += 32) {
    v16h a0 = load_frag(A, K, m0,      k0, lane);
    v16h a1 = load_frag(A, K, m0 + 16, k0, lane);
#pragma unroll
    for (int t = 0; t < 4; ++t) {
      v16h b = load_frag(W, K, n0 + 16 * t, k0, lane);
      acc[0][t] = WMMA_F16(a0, b, acc[0][t]);
      acc[1][t] = WMMA_F16(a1, b, acc[1][t]);
    }
  }
  int g = lane >> 4, c = lane & 15;
#pragma unroll
  for (int u = 0; u < 2; ++u) {
#pragma unroll
    for (int t = 0; t < 4; ++t) {
#pragma unroll
      for (int r = 0; r < 8; ++r) {
        int m = m0 + 16 * u + r + 8 * g;
        int n = n0 + 16 * t + c;
        out[(size_t)m * N + n] = acc[u][t][r] + bias[n];
      }
    }
  }
}

// ---------------------------------------------------------------------------
extern "C" void kernel_launch(void* const* d_in, const int* in_sizes, int n_in,
                              void* d_out, int out_size, void* d_ws, size_t ws_size,
                              hipStream_t stream) {
  const float* x   = (const float*)d_in[0];
  const float* ctx = (const float*)d_in[1];
  const float* Wq  = (const float*)d_in[2];
  const float* Wk  = (const float*)d_in[3];
  const float* Wv  = (const float*)d_in[4];
  const float* Wo  = (const float*)d_in[5];
  const float* bo  = (const float*)d_in[6];

  const size_t NE_BIG = (size_t)4 * 2048 * 1024;  // 8,388,608 elems
  const size_t NE_W   = (size_t)1024 * 1024;

  _Float16* ws  = (_Float16*)d_ws;
  _Float16* Xh  = ws;                 // f16 x
  _Float16* Ch  = Xh  + NE_BIG;       // f16 context
  _Float16* Wqh = Ch  + NE_BIG;
  _Float16* Wkh = Wqh + NE_W;
  _Float16* Wvh = Wkh + NE_W;
  _Float16* Woh = Wvh + NE_W;
  _Float16* Qp  = Woh + NE_W;         // (b,h,i,d), pre-scaled by 1/8
  _Float16* Kp  = Qp  + NE_BIG;       // (b,h,j,d)
  _Float16* Vt  = Kp  + NE_BIG;       // (b,h,d,j)
  _Float16* AO  = Vt  + NE_BIG;       // (b,i,h*64+d)

  const int nb = (int)NE_BIG, nw = (int)NE_W;
  cvt_f32_to_f16<<<(nb + 255) / 256, 256, 0, stream>>>(x,   Xh,  nb);
  cvt_f32_to_f16<<<(nb + 255) / 256, 256, 0, stream>>>(ctx, Ch,  nb);
  cvt_f32_to_f16<<<(nw + 255) / 256, 256, 0, stream>>>(Wq,  Wqh, nw);
  cvt_f32_to_f16<<<(nw + 255) / 256, 256, 0, stream>>>(Wk,  Wkh, nw);
  cvt_f32_to_f16<<<(nw + 255) / 256, 256, 0, stream>>>(Wv,  Wvh, nw);
  cvt_f32_to_f16<<<(nw + 255) / 256, 256, 0, stream>>>(Wo,  Woh, nw);

  // M = 8192 rows, N = 1024 cols; block = 8 waves * 32 rows, 64 cols
  dim3 ggrid(32, 16);
  proj_gemm<<<ggrid, 256, 0, stream>>>(Xh, Wqh, Qp, 0, 2048, 0.125f); // Q*SCALE
  proj_gemm<<<ggrid, 256, 0, stream>>>(Ch, Wkh, Kp, 0, 2048, 1.0f);
  proj_gemm<<<ggrid, 256, 0, stream>>>(Ch, Wvh, Vt, 1, 2048, 1.0f);

  // 4*16*(2048/32) = 4096 query tiles, 8 waves per block
  attn_kernel<<<512, 256, 0, stream>>>(Qp, Kp, Vt, AO);

  out_gemm<<<ggrid, 256, 0, stream>>>(AO, Woh, bo, (float*)d_out);
}
